// ExtractKeyframes_10806137717417
// MI455X (gfx1250) — compile-verified, matrix operand order checked by reading, not run on hardware
//
#include <hip/hip_runtime.h>
#include <hip/hip_bf16.h>

typedef __attribute__((ext_vector_type(16))) __bf16 v16bf;
typedef __attribute__((ext_vector_type(8)))  float  v8f;
typedef __attribute__((ext_vector_type(4)))  float  f4;

#define V_DIM 128
#define F_DIM 64
#define T_DIM 256
#define D_DIM 512
#define KSEL  8

// One workgroup = (video v, 16-row text tile t0). 4 waves, each computing a
// 16(t) x 16(f) WMMA tile; together they cover all F=64 frames. Fused:
// bf16 WMMA sims -> LDS -> top-8 select -> pooled gather-sum + index broadcast.
__global__ __launch_bounds__(128) void ExtractKeyframes_fused_kernel(
    const float* __restrict__ text,    // [T=256][D=512]
    const float* __restrict__ video,   // [V=128][F=64][D=512]
    float* __restrict__ pooled,        // [V][T][D]
    float* __restrict__ out2)          // [V][K=8][D][T]
{
    __shared__ float simsLDS[16][F_DIM];   // 4 KB: 16 text rows x 64 frames
    __shared__ int   idxLDS[16][KSEL];

    const int v    = blockIdx.x;
    const int t0   = blockIdx.y << 4;      // text tile base
    const int tid  = threadIdx.x;
    const int wave = tid >> 5;
    const int lane = tid & 31;
    const int l15  = lane & 15;
    const int kbase = (lane < 16) ? 0 : 8; // per-ISA 16-bit A/B lane K layout
    const int f0   = wave << 4;            // frame tile base for this wave

    // A: text row (M = t0+l15), contiguous along d.
    // B: frame row (N = f0+l15), also contiguous along d (column of B = K dim).
    const float* arow = text  + (size_t)(t0 + l15) * D_DIM + kbase;
    const float* brow = video + ((size_t)(v * F_DIM + f0 + l15)) * D_DIM + kbase;

    v8f acc = {};
    for (int kk = 0; kk < D_DIM; kk += 32) {
        // Two 8-float runs per matrix: K in [kbase, kbase+8) and [kbase+16, kbase+24)
        float af[16], bf[16];
        *(f4*)(af + 0)  = *(const f4*)(arow + kk + 0);
        *(f4*)(af + 4)  = *(const f4*)(arow + kk + 4);
        *(f4*)(af + 8)  = *(const f4*)(arow + kk + 16);
        *(f4*)(af + 12) = *(const f4*)(arow + kk + 20);
        *(f4*)(bf + 0)  = *(const f4*)(brow + kk + 0);
        *(f4*)(bf + 4)  = *(const f4*)(brow + kk + 4);
        *(f4*)(bf + 8)  = *(const f4*)(brow + kk + 16);
        *(f4*)(bf + 12) = *(const f4*)(brow + kk + 20);

        v16bf A, B;
#pragma unroll
        for (int i = 0; i < 16; ++i) {
            A[i] = (__bf16)af[i];
            B[i] = (__bf16)bf[i];
        }
        // D = A(16x32 bf16) x B(32x16 bf16) + C(f32)
        acc = __builtin_amdgcn_wmma_f32_16x16x32_bf16(
            /*neg_a=*/false, A, /*neg_b=*/false, B,
            /*c_mod=*/(short)0, acc, /*reuse_a=*/false, /*reuse_b=*/false);
    }

    // C layout: VGPR r -> M = r + 8*(lane>=16), N = lane&15
#pragma unroll
    for (int r = 0; r < 8; ++r) {
        int tt = r + ((lane >> 4) << 3);
        simsLDS[tt][f0 + l15] = acc[r];
    }
    __syncthreads();

    // Top-8 per text row: strict '>' keeps lowest index on ties (lax.top_k).
    if (tid < 16) {
        unsigned long long used = 0ull;
        for (int k = 0; k < KSEL; ++k) {
            float best = -3.402823466e38f;
            int   bi   = 0;
            for (int f = 0; f < F_DIM; ++f) {
                float s = simsLDS[tid][f];
                if (!((used >> f) & 1ull) && s > best) { best = s; bi = f; }
            }
            used |= (1ull << bi);
            idxLDS[tid][k] = bi;
        }
    }
    __syncthreads();

    // pooled[v][t0+tt][d] = sum over top-8 frames of video[v][idx][d]
    const float* vbase = video + (size_t)v * F_DIM * D_DIM;
    for (int tt = 0; tt < 16; ++tt) {
        int fidx[KSEL];
#pragma unroll
        for (int k = 0; k < KSEL; ++k) fidx[k] = idxLDS[tt][k];
        for (int d = tid; d < D_DIM; d += 128) {
            float s = 0.0f;
#pragma unroll
            for (int k = 0; k < KSEL; ++k)
                s += vbase[(size_t)fidx[k] * D_DIM + d];
            pooled[((size_t)(v * T_DIM) + t0 + tt) * D_DIM + d] = s;
        }
    }

    // out2[v][k][d][t] = (float)idx[v][t][k]  -- broadcast along d.
    // lanes 0..15 cover 16 consecutive t (64B contiguous stores).
    {
        const int t    = tid & 15;
        const int dgrp = tid >> 4;  // 0..7
#pragma unroll
        for (int k = 0; k < KSEL; ++k) {
            float val = (float)idxLDS[t][k];
            size_t base = ((size_t)v * KSEL + k) * D_DIM * T_DIM;
            for (int d = dgrp; d < D_DIM; d += 8) {
                out2[base + (size_t)d * T_DIM + t0 + t] = val;
            }
        }
    }
}

extern "C" void kernel_launch(void* const* d_in, const int* in_sizes, int n_in,
                              void* d_out, int out_size, void* d_ws, size_t ws_size,
                              hipStream_t stream) {
    (void)in_sizes; (void)n_in; (void)out_size; (void)d_ws; (void)ws_size;
    const float* text  = (const float*)d_in[0];   // [256*512]
    const float* video = (const float*)d_in[1];   // [128*64*512]
    float* pooled = (float*)d_out;                                  // [V*T*D]
    float* out2   = (float*)d_out + (size_t)V_DIM * T_DIM * D_DIM;  // [V*K*D*T]

    dim3 grid(V_DIM, T_DIM / 16);   // 128 x 16 = 2048 workgroups
    ExtractKeyframes_fused_kernel<<<grid, 128, 0, stream>>>(text, video, pooled, out2);
}
